// PDENetLight_26800595927155
// MI455X (gfx1250) — matrix operand order repaired and checked by reading, not compiled, not run on hardware
//
#include <hip/hip_runtime.h>

// ---------------------------------------------------------------------------
// PDENetLight fused inference for gfx1250 (MI455X, wave32, WMMA).
//   out(1,N) = W @ S3 + b  with two GRU-style layers, HID=512, N=65536.
// Compute-bound: 8 GEMMs (512x512)@(512x65536) ~ 275 GFLOP vs ~9 MB of
// mandatory HBM traffic (~0.4us @ 23.3 TB/s). Strategy: per-WG tile of NT=32
// columns; all activations resident in LDS as f16 (column-major, padded
// stride); GEMMs via v_wmma_f32_16x16x32_f16 (f16 in, fp32 accum); U@x+b
// fused into accumulator init; tanh/gating fused into epilogues with packed
// f16 stores; weights pre-packed to f16 A-fragment layout in d_ws (L2-hot).
// ---------------------------------------------------------------------------

#define DIM_IN 3
#define HIDN   512
#define NTOT   65536
#define NT     32                 // columns of N per workgroup
#define CST    528                // padded LDS column stride in halfs (32B multiple)
#define KTILE  32                 // K per WMMA
#define NKT    (HIDN / KTILE)     // 16 K-steps
#define NMT    (HIDN / 16)        // 32 M-tiles

typedef __attribute__((ext_vector_type(16))) _Float16 v16h;
typedef __attribute__((ext_vector_type(2)))  _Float16 v2h;
typedef __attribute__((ext_vector_type(8)))  float    v8f;

// ---------------------------------------------------------------------------
// Weight pack: fp32 row-major (512,512) -> f16 16x32 A fragments in this file's own
// contiguous-K order (B reads the same). One thread = one lane of one fragment; writes
// its 16 contiguous halfs, so the GEMM loads are 2x global_load_b128 / lane.
// Fragment offset (halfs): ((mt*NKT + kt)*32 + lane)*16.
// Lane<16 holds M=lane,   K = {0..15} (+kt*32)
// Lane>=16 holds M=lane-16, K = {16..31} (+kt*32)
// ---------------------------------------------------------------------------
struct PackArgs { const float* w[8]; _Float16* dst; };

__global__ __launch_bounds__(256) void pack_weights_kernel(PackArgs pa) {
  int gid  = blockIdx.x * 256 + threadIdx.x;     // 8*32*16*32 = 131072 threads
  int lane = gid & 31;
  int kt   = (gid >> 5) & (NKT - 1);
  int mt   = (gid >> 9) & (NMT - 1);
  int mat  = gid >> 14;
  const float* W = pa.w[mat];
  _Float16* dst = pa.dst + (size_t)mat * HIDN * HIDN
                + ((size_t)(mt * NKT + kt) * 32 + lane) * 16;
  int M  = mt * 16 + (lane & 15);
  int kb = kt * KTILE + ((lane < 16) ? 0 : 16);
#pragma unroll
  for (int h = 0; h < 16; ++h) {
    int k = kb + h;
    dst[h] = (_Float16)W[(size_t)M * HIDN + k];
  }
}

// ---------------------------------------------------------------------------
// Fused network kernel
// ---------------------------------------------------------------------------
struct MainArgs {
  const float* x;                    // (3,N)
  const float* W1; const float* b1;  // (512,3),(512,1)
  const float* U[8];                 // Uz1,Ug1,Ur1,Uh1,Uz2,Ug2,Ur2,Uh2 (512,3)
  const float* bias[8];              // bz1,bg1,br1,bh1,bz2,...        (512,1)
  const _Float16* packed;            // 8 packed (512,512) f16 matrices
  const float* Wout; const float* bout;
  float* out;
};

__device__ __forceinline__ float fast_tanh(float v) {
  // tanh(v) = (e^{2v}-1)/(e^{2v}+1), e^{2v} = 2^{2v*log2(e)}
  v = fminf(fmaxf(v, -15.0f), 15.0f);
  float e = __builtin_amdgcn_exp2f(v * 2.88539008177793f);
  return (e - 1.0f) * __builtin_amdgcn_rcpf(e + 1.0f);
}

// One 512xNT gate GEMM for this wave: acc = U@x + b ; acc += Wpack @ Bin ;
// epilogue per MODE: 0: Dst=tanh(acc)   1: Dst=Saux*tanh(acc)
//                    2: Dst=(1-Gaux)*tanh(acc) + Zaux*Saux   (Dst may alias Zaux;
//                       safe: every element is read+written by its owning thread)
template <int MODE>
__device__ __forceinline__ void gate_gemm(
    const _Float16* __restrict__ Apack,
    const float* __restrict__ U, const float* __restrict__ bias,
    const float* __restrict__ xs,
    const _Float16* __restrict__ Bin, _Float16* Dst,
    const _Float16* Saux, const _Float16* Zaux, const _Float16* Gaux,
    int lane, int wave)
{
  const int hi = lane >> 4;       // which half of the lanes
  const int ln = lane & 15;
  v8f acc[4][2];

  // ---- accumulator init with the K=3 input term (U@x + b) ----
#pragma unroll
  for (int mt = 0; mt < 4; ++mt) {
    const int mbase = wave * 64 + mt * 16 + hi * 8;
#pragma unroll
    for (int v = 0; v < 8; ++v) {
      const int m = mbase + v;
      const float u0 = U[m * 3 + 0], u1 = U[m * 3 + 1], u2 = U[m * 3 + 2];
      const float bb = bias[m];
#pragma unroll
      for (int nt = 0; nt < 2; ++nt) {
        const int n = nt * 16 + ln;
        acc[mt][nt][v] = bb + u0 * xs[n] + u1 * xs[NT + n] + u2 * xs[2 * NT + n];
      }
    }
  }

  // ---- K loop: 16 steps x (4x2) WMMA tiles ----
  const _Float16* aptr = Apack + ((size_t)(wave * 4) * NKT * 32 + lane) * 16;
  for (int kt = 0; kt < NKT; ++kt) {
    v16h a[4];
#pragma unroll
    for (int mt = 0; mt < 4; ++mt)
      a[mt] = *(const v16h*)(aptr + ((size_t)(mt * NKT + kt) * 32) * 16);
    if (kt + 1 < NKT)   // keep next weight slab flowing through L2/L0
      __builtin_prefetch((const void*)(aptr + ((size_t)(kt + 1) * 32) * 16), 0, 1);

    v16h b[2];
#pragma unroll
    for (int nt = 0; nt < 2; ++nt)   // B 32x16 frag: lane ln = col, hi -> K half
      b[nt] = *(const v16h*)(Bin + (size_t)(nt * 16 + ln) * CST + kt * KTILE + hi * 16);

#pragma unroll
    for (int mt = 0; mt < 4; ++mt)
#pragma unroll
      for (int nt = 0; nt < 2; ++nt)
        acc[mt][nt] = __builtin_amdgcn_wmma_f32_16x16x32_f16(
            false, a[mt], false, b[nt], (short)0, acc[mt][nt], false, false);
  }

  // ---- fused elementwise epilogue (C/D layout: row = v + hi*8, col = ln) ----
  // consecutive v => consecutive M => contiguous halfs in LDS: pack 2 results
  // per ds_store_b32.
#pragma unroll
  for (int mt = 0; mt < 4; ++mt) {
#pragma unroll
    for (int nt = 0; nt < 2; ++nt) {
#pragma unroll
      for (int v = 0; v < 8; v += 2) {
        const int m = wave * 64 + mt * 16 + hi * 8 + v;
        const int n = nt * 16 + ln;
        const int idx = n * CST + m;       // even -> 4B aligned
        float t0 = fast_tanh(acc[mt][nt][v]);
        float t1 = fast_tanh(acc[mt][nt][v + 1]);
        if (MODE == 1) {
          const v2h s2 = *(const v2h*)(Saux + idx);
          t0 *= (float)s2[0];
          t1 *= (float)s2[1];
        } else if (MODE == 2) {
          const v2h g2 = *(const v2h*)(Gaux + idx);
          const v2h z2 = *(const v2h*)(Zaux + idx);
          const v2h s2 = *(const v2h*)(Saux + idx);
          t0 = (1.0f - (float)g2[0]) * t0 + (float)z2[0] * (float)s2[0];
          t1 = (1.0f - (float)g2[1]) * t1 + (float)z2[1] * (float)s2[1];
        }
        v2h pk;
        pk[0] = (_Float16)t0;
        pk[1] = (_Float16)t1;
        *(v2h*)(Dst + idx) = pk;
      }
    }
  }
}

__global__ __launch_bounds__(256, 2) void fused_gru_pde_kernel(MainArgs A) {
  extern __shared__ char smem_raw[];
  _Float16* buf0 = (_Float16*)smem_raw;        // S
  _Float16* buf1 = buf0 + NT * CST;            // Z
  _Float16* buf2 = buf1 + NT * CST;            // G
  _Float16* buf3 = buf2 + NT * CST;            // P = S*R
  float* xs  = (float*)(buf3 + NT * CST);      // 3 x NT
  float* red = xs + DIM_IN * NT;               // NT x 8 head partials

  const int tid  = threadIdx.x;
  const int lane = tid & 31;
  const int wave = tid >> 5;
  const int col0 = blockIdx.x * NT;

  // stage x tile
  for (int i = tid; i < DIM_IN * NT; i += 256) {
    int d = i / NT, n = i - d * NT;
    xs[i] = A.x[(size_t)d * NTOT + col0 + n];
  }
  __syncthreads();

  // S1 = tanh(W1@x + b1), K=3 -> plain VALU; pack pairs along m for b32 stores
  for (int i = tid; i < (HIDN / 2) * NT; i += 256) {
    int mh = i & (HIDN / 2 - 1);               // m/2
    int n  = i / (HIDN / 2);
    int m  = mh * 2;
    float t0 = A.b1[m]     + A.W1[m * 3 + 0] * xs[n]
                           + A.W1[m * 3 + 1] * xs[NT + n]
                           + A.W1[m * 3 + 2] * xs[2 * NT + n];
    float t1 = A.b1[m + 1] + A.W1[m * 3 + 3] * xs[n]
                           + A.W1[m * 3 + 4] * xs[NT + n]
                           + A.W1[m * 3 + 5] * xs[2 * NT + n];
    v2h pk;
    pk[0] = (_Float16)fast_tanh(t0);
    pk[1] = (_Float16)fast_tanh(t1);
    *(v2h*)(buf0 + n * CST + m) = pk;
  }
  __syncthreads();

  _Float16 *S = buf0, *Z = buf1, *G = buf2, *P = buf3;

  for (int layer = 0; layer < 2; ++layer) {
    const int g0 = layer * 4;
    const _Float16* pk = A.packed;
    // Z = tanh(Uz@x + Wz@S + bz)
    gate_gemm<0>(pk + (size_t)(g0 + 0) * HIDN * HIDN, A.U[g0 + 0], A.bias[g0 + 0],
                 xs, S, Z, nullptr, nullptr, nullptr, lane, wave);
    __syncthreads();
    // G = tanh(Ug@x + Wg@S + bg)
    gate_gemm<0>(pk + (size_t)(g0 + 1) * HIDN * HIDN, A.U[g0 + 1], A.bias[g0 + 1],
                 xs, S, G, nullptr, nullptr, nullptr, lane, wave);
    __syncthreads();
    // P = S * tanh(Ur@x + Wr@S + br)
    gate_gemm<1>(pk + (size_t)(g0 + 2) * HIDN * HIDN, A.U[g0 + 2], A.bias[g0 + 2],
                 xs, S, P, S, nullptr, nullptr, lane, wave);
    __syncthreads();
    // Snew = (1-G)*tanh(Uh@x + Wh@P + bh) + Z*S  -> written over Z buffer
    gate_gemm<2>(pk + (size_t)(g0 + 3) * HIDN * HIDN, A.U[g0 + 3], A.bias[g0 + 3],
                 xs, P, Z, S, Z, G, lane, wave);
    __syncthreads();
    _Float16* tsw = S; S = Z; Z = tsw;     // S <- Snew
  }

  // head: out[n] = b + sum_m W[m]*S3[m,n]; 8 threads per column, paired loads
  const int n = tid & (NT - 1);
  const int chunk = tid >> 5;
  float partial = 0.0f;
  const v2h* colp2 = (const v2h*)(S + (size_t)n * CST + chunk * 64);
  const float* wp = A.Wout + chunk * 64;
#pragma unroll 8
  for (int m2 = 0; m2 < 32; ++m2) {
    const v2h hv = colp2[m2];
    partial += wp[2 * m2] * (float)hv[0] + wp[2 * m2 + 1] * (float)hv[1];
  }
  red[n * 8 + chunk] = partial;
  __syncthreads();
  if (tid < NT) {
    float s = A.bout[0];
#pragma unroll
    for (int c = 0; c < 8; ++c) s += red[tid * 8 + c];
    A.out[col0 + tid] = s;
  }
}

// ---------------------------------------------------------------------------
extern "C" void kernel_launch(void* const* d_in, const int* in_sizes, int n_in,
                              void* d_out, int out_size, void* d_ws, size_t ws_size,
                              hipStream_t stream) {
  (void)in_sizes; (void)n_in; (void)out_size; (void)ws_size;
  // input order: 0:x 1:W1 2:b1 then per gate (U,W,b): z1,g1,r1,h1,z2,g2,r2,h2,
  // 27:W 28:b
  const int Uidx[8] = {3, 6, 9, 12, 15, 18, 21, 24};
  const int Widx[8] = {4, 7, 10, 13, 16, 19, 22, 25};
  const int Bidx[8] = {5, 8, 11, 14, 17, 20, 23, 26};

  _Float16* packed = (_Float16*)d_ws;   // 8 * 512 * 512 * 2B = 4 MB

  PackArgs pa;
  for (int i = 0; i < 8; ++i) pa.w[i] = (const float*)d_in[Widx[i]];
  pa.dst = packed;
  pack_weights_kernel<<<512, 256, 0, stream>>>(pa);

  MainArgs ma;
  ma.x  = (const float*)d_in[0];
  ma.W1 = (const float*)d_in[1];
  ma.b1 = (const float*)d_in[2];
  for (int i = 0; i < 8; ++i) {
    ma.U[i]    = (const float*)d_in[Uidx[i]];
    ma.bias[i] = (const float*)d_in[Bidx[i]];
  }
  ma.packed = packed;
  ma.Wout = (const float*)d_in[27];
  ma.bout = (const float*)d_in[28];
  ma.out  = (float*)d_out;

  size_t smem = (size_t)4 * NT * CST * sizeof(_Float16)
              + (size_t)DIM_IN * NT * sizeof(float)
              + (size_t)NT * 8 * sizeof(float);   // ~136.6 KB of the 320 KB WGP LDS
  hipFuncSetAttribute(reinterpret_cast<const void*>(fused_gru_pde_kernel),
                      hipFuncAttributeMaxDynamicSharedMemorySize, (int)smem);
  fused_gru_pde_kernel<<<NTOT / NT, 256, smem, stream>>>(ma);
}
